// Decoder_90907277787574
// MI455X (gfx1250) — matrix operand order repaired and checked by reading, not compiled, not run on hardware
//
#include <hip/hip_runtime.h>
#include <hip/hip_bf16.h>
#include <math.h>

// ---------------------------------------------------------------------------
// CDNA5 (gfx1250) WMMA types
// ---------------------------------------------------------------------------
typedef __attribute__((ext_vector_type(16))) __bf16 bf16x16;
typedef __attribute__((ext_vector_type(8)))  __bf16 bf16x8;
typedef __attribute__((ext_vector_type(8)))  float  floatx8;

#define WMMA_BF16(A, B, C) \
    __builtin_amdgcn_wmma_f32_16x16x32_bf16(false, (A), false, (B), (short)0, (C), false, false)

// Problem constants (from reference)
static const int B_  = 256;
static const int H_  = 1024;
static const int G_  = 3072;    // 3*H
static const int V_  = 32000;
static const int T_  = 20;      // MAXLEN
static const int L_  = 4;       // NLAYERS

// ---------------------------------------------------------------------------
// Fragment helpers (wave32, v_wmma_f32_16x16x32_bf16)
//
// A (16x32 bf16), row-major bf16 source:
//   lane L: row = lane&15 ; elems 0..15 -> K = k0 + (lane>>4)*16 + e
//   => two 16B contiguous loads per lane.
// B (32x16 bf16), source W[N][K] row-major (B[k][n] = W[n][k]):
//   lane L: col = lane&15 ; elems 0..15 -> K = k0 + (lane>>4)*16 + e
//   => 32B contiguous per lane.
// C/D (16x16 f32): VGPR v, lane L: N = lane&15 ; M = v + (lane>>4)*8
// ---------------------------------------------------------------------------
__device__ __forceinline__ bf16x16 load_a_bf16(const __bf16* __restrict__ rowPtr,
                                               int k0, int lane) {
    const __bf16* p = rowPtr + k0 + ((lane >> 4) << 4);
    bf16x8 lo = *(const bf16x8*)p;
    bf16x8 hi = *(const bf16x8*)(p + 8);
    bf16x16 a;
#pragma unroll
    for (int e = 0; e < 8; ++e) { a[e] = lo[e]; a[e + 8] = hi[e]; }
    return a;
}

__device__ __forceinline__ bf16x16 load_b_bf16(const __bf16* __restrict__ w,
                                               int ldk, int col, int k0, int lane) {
    const __bf16* p = w + (size_t)col * ldk + k0 + ((lane >> 4) << 4);
    return *(const bf16x16*)p;   // 32B, 32B-aligned -> 2x b128 loads
}

// ---------------------------------------------------------------------------
// Weight conversion: f32 -> bf16 (once per launch; amortized over 20 steps)
// ---------------------------------------------------------------------------
__global__ void f32_to_bf16_kernel(const float* __restrict__ src,
                                   __bf16* __restrict__ dst, size_t n) {
    size_t i = (size_t)blockIdx.x * blockDim.x + threadIdx.x;
    size_t stride = (size_t)gridDim.x * blockDim.x;
    for (; i < n; i += stride) dst[i] = (__bf16)src[i];
}

// ---------------------------------------------------------------------------
// Embedding gather + ReLU, written directly as bf16 GEMM input:
//   x_bf[b][j] = bf16(relu(emb[tok[b]][j])) ; t==0 -> tok=0
// ---------------------------------------------------------------------------
__global__ void embed_relu_kernel(const float* __restrict__ emb,
                                  const int* __restrict__ tok, int t,
                                  __bf16* __restrict__ x_bf) {
    int idx = blockIdx.x * blockDim.x + threadIdx.x;   // B*H threads
    int b = idx >> 10;
    int j = idx & 1023;
    int tk = (t == 0) ? 0 : tok[b];
    float v = emb[(size_t)tk * H_ + j];
    x_bf[idx] = (__bf16)(v > 0.0f ? v : 0.0f);
}

// ---------------------------------------------------------------------------
// Generic GEMM:  C[256 x N] = A[256 x 1024] @ W[N x 1024]^T (+ bias)
// Each wave computes a 16x64 strip: one A-fragment feeds 4 WMMAs per K-step.
// Grid: (16 * N/64) wave-strips / 8 waves per block.
// ---------------------------------------------------------------------------
__global__ __launch_bounds__(256)
void gemm_bf16_kernel(const __bf16* __restrict__ A,     // [256,1024]
                      const __bf16* __restrict__ W,     // [N,1024]
                      const float* __restrict__ bias,   // [N] or nullptr
                      float* __restrict__ C,            // [256,N]
                      int N) {
    int wave = threadIdx.x >> 5;
    int lane = threadIdx.x & 31;
    int strip = blockIdx.x * 8 + wave;
    int nGroups = N >> 6;                 // N/64
    int mt = strip / nGroups;
    int ng = strip - mt * nGroups;
    int mBase = mt * 16, nBase = ng * 64;

    const __bf16* aRow = A + (size_t)(mBase + (lane & 15)) * H_;
    int colBase = nBase + (lane & 15);

    floatx8 c0 = {}, c1 = {}, c2 = {}, c3 = {};
    for (int k0 = 0; k0 < H_; k0 += 32) {
        if (k0 + 64 < H_) {   // weight stream lookahead -> global_prefetch_b8
            __builtin_prefetch(W + (size_t)colBase * H_ + k0 + 64, 0, 1);
            __builtin_prefetch(W + (size_t)(colBase + 32) * H_ + k0 + 64, 0, 1);
        }
        bf16x16 a  = load_a_bf16(aRow, k0, lane);
        bf16x16 b0 = load_b_bf16(W, H_, colBase,      k0, lane);
        bf16x16 b1 = load_b_bf16(W, H_, colBase + 16, k0, lane);
        bf16x16 b2 = load_b_bf16(W, H_, colBase + 32, k0, lane);
        bf16x16 b3 = load_b_bf16(W, H_, colBase + 48, k0, lane);
        c0 = WMMA_BF16(a, b0, c0);
        c1 = WMMA_BF16(a, b1, c1);
        c2 = WMMA_BF16(a, b2, c2);
        c3 = WMMA_BF16(a, b3, c3);
    }

    int m0 = mBase + ((lane >> 4) << 3);
    int n  = nBase + (lane & 15);
    floatx8 acc[4] = {c0, c1, c2, c3};
#pragma unroll
    for (int q = 0; q < 4; ++q) {
        int nq = n + q * 16;
        float bv = bias ? bias[nq] : 0.0f;
#pragma unroll
        for (int v = 0; v < 8; ++v)
            C[(size_t)(m0 + v) * N + nq] = acc[q][v] + bv;
    }
}

// ---------------------------------------------------------------------------
// GRU pointwise gates (PyTorch order r,z,n); updates h (fp32 carry) and
// emits bf16 copies of h_new for the next step's GEMMs (h_bf, x_bf).
// ---------------------------------------------------------------------------
__global__ void gru_pointwise_kernel(const float* __restrict__ gi,
                                     const float* __restrict__ gh,
                                     const float* __restrict__ bih,  // [3072]
                                     const float* __restrict__ bhh,  // [3072]
                                     float* __restrict__ h,          // [256,1024]
                                     __bf16* __restrict__ h_bf,      // [256,1024]
                                     __bf16* __restrict__ x_bf) {    // [256,1024]
    int idx = blockIdx.x * blockDim.x + threadIdx.x;   // B*H
    int b = idx >> 10;
    int j = idx & 1023;
    size_t g = (size_t)b * G_;
    float ir  = gi[g + j]        + bih[j];
    float iz  = gi[g + 1024 + j] + bih[1024 + j];
    float in_ = gi[g + 2048 + j] + bih[2048 + j];
    float hr  = gh[g + j]        + bhh[j];
    float hz  = gh[g + 1024 + j] + bhh[1024 + j];
    float hn  = gh[g + 2048 + j] + bhh[2048 + j];
    float r = 1.0f / (1.0f + expf(-(ir + hr)));
    float z = 1.0f / (1.0f + expf(-(iz + hz)));
    float n = tanhf(in_ + r * hn);
    float ho = h[idx];
    float hnew = (1.0f - z) * n + z * ho;
    h[idx] = hnew;
    __bf16 hb = (__bf16)hnew;
    h_bf[idx] = hb;     // this layer's hidden, next step
    x_bf[idx] = hb;     // next layer's input, this step
}

// ---------------------------------------------------------------------------
// Per-row argmax (greedy token; first-max tie break) + row max for softmax.
// ---------------------------------------------------------------------------
__global__ void argmax_rows_kernel(const float* __restrict__ logits,
                                   int* __restrict__ tok,
                                   float* __restrict__ rowmax) {
    __shared__ float sv[256];
    __shared__ int   si[256];
    int b = blockIdx.x;
    const float* row = logits + (size_t)b * V_;
    float best = -3.402823466e38f;
    int bidx = 0x7fffffff;
    for (int j = threadIdx.x; j < V_; j += 256) {
        float v = row[j];
        if (v > best || (v == best && j < bidx)) { best = v; bidx = j; }
    }
    sv[threadIdx.x] = best;
    si[threadIdx.x] = bidx;
    __syncthreads();
    for (int s = 128; s > 0; s >>= 1) {
        if (threadIdx.x < s) {
            float v2 = sv[threadIdx.x + s];
            int   i2 = si[threadIdx.x + s];
            if (v2 > sv[threadIdx.x] ||
                (v2 == sv[threadIdx.x] && i2 < si[threadIdx.x])) {
                sv[threadIdx.x] = v2;
                si[threadIdx.x] = i2;
            }
        }
        __syncthreads();
    }
    if (threadIdx.x == 0) { tok[b] = si[0]; rowmax[b] = sv[0]; }
}

// ---------------------------------------------------------------------------
// Per-row logsumexp + write log_probs slice for step t: out[b][t][:]
// ---------------------------------------------------------------------------
__global__ void logsoftmax_write_kernel(const float* __restrict__ logits,
                                        const float* __restrict__ rowmax,
                                        float* __restrict__ out, int t) {
    __shared__ float ss[256];
    int b = blockIdx.x;
    const float* row = logits + (size_t)b * V_;
    float m = rowmax[b];
    float s = 0.0f;
    for (int j = threadIdx.x; j < V_; j += 256) s += expf(row[j] - m);
    ss[threadIdx.x] = s;
    __syncthreads();
    for (int k = 128; k > 0; k >>= 1) {
        if (threadIdx.x < k) ss[threadIdx.x] += ss[threadIdx.x + k];
        __syncthreads();
    }
    float lse = m + logf(ss[0]);
    float* o = out + ((size_t)b * T_ + t) * V_;
    for (int j = threadIdx.x; j < V_; j += 256) o[j] = row[j] - lse;
}

// ---------------------------------------------------------------------------
// Host orchestration
// ---------------------------------------------------------------------------
extern "C" void kernel_launch(void* const* d_in, const int* in_sizes, int n_in,
                              void* d_out, int out_size, void* d_ws, size_t ws_size,
                              hipStream_t stream) {
    (void)in_sizes; (void)n_in; (void)out_size; (void)ws_size;
    // Inputs (setup_inputs order):
    // 0 encoder_outputs (unused), 1 encoder_hidden [4,256,1024], 2 emb_table,
    // 3 gru_w_ih [4,3072,1024], 4 gru_w_hh, 5 gru_b_ih [4,3072], 6 gru_b_hh,
    // 7 out_w [32000,1024], 8 out_b [32000], 9..12 l1/l2 (dead code in ref).
    const float* enc_hid = (const float*)d_in[1];
    const float* emb     = (const float*)d_in[2];
    const float* w_ih    = (const float*)d_in[3];
    const float* w_hh    = (const float*)d_in[4];
    const float* b_ih    = (const float*)d_in[5];
    const float* b_hh    = (const float*)d_in[6];
    const float* out_w   = (const float*)d_in[7];
    const float* out_b   = (const float*)d_in[8];
    float* out = (float*)d_out;

    // Workspace layout (~162 MB)
    char* ws = (char*)d_ws;
    size_t off = 0;
    const size_t nGruW = (size_t)L_ * G_ * H_;       // 12,582,912
    const size_t nOutW = (size_t)V_ * H_;            // 32,768,000
    const size_t nBH   = (size_t)B_ * H_;            // 262,144
    __bf16* wih_bf  = (__bf16*)(ws + off); off += nGruW * 2;
    __bf16* whh_bf  = (__bf16*)(ws + off); off += nGruW * 2;
    __bf16* outw_bf = (__bf16*)(ws + off); off += nOutW * 2;
    float*  hstate  = (float*)(ws + off);  off += (size_t)L_ * nBH * 4;
    __bf16* h_bf    = (__bf16*)(ws + off); off += (size_t)L_ * nBH * 2;
    __bf16* x_bf    = (__bf16*)(ws + off); off += nBH * 2;
    float*  gi      = (float*)(ws + off);  off += (size_t)B_ * G_ * 4;
    float*  gh      = (float*)(ws + off);  off += (size_t)B_ * G_ * 4;
    float*  logits  = (float*)(ws + off);  off += (size_t)B_ * V_ * 4;
    int*    tok     = (int*)(ws + off);    off += B_ * 4;
    float*  rowmax  = (float*)(ws + off);  off += B_ * 4;

    // One-time (per launch) conversions to bf16
    f32_to_bf16_kernel<<<2048, 256, 0, stream>>>(w_ih,  wih_bf,  nGruW);
    f32_to_bf16_kernel<<<2048, 256, 0, stream>>>(w_hh,  whh_bf,  nGruW);
    f32_to_bf16_kernel<<<4096, 256, 0, stream>>>(out_w, outw_bf, nOutW);
    f32_to_bf16_kernel<<<1024, 256, 0, stream>>>(enc_hid, h_bf, (size_t)L_ * nBH);

    // h <- encoder_hidden (fp32 carry)
    hipMemcpyAsync(hstate, enc_hid, (size_t)L_ * nBH * sizeof(float),
                   hipMemcpyDeviceToDevice, stream);

    const int bhBlocks     = (int)(nBH / 256);        // 1024
    const int gateBlocks   = (16 * (G_ / 64)) / 8;    // 96
    const int logitsBlocks = (16 * (V_ / 64)) / 8;    // 1000

    for (int t = 0; t < T_; ++t) {
        embed_relu_kernel<<<bhBlocks, 256, 0, stream>>>(emb, tok, t, x_bf);
        for (int l = 0; l < L_; ++l) {
            const __bf16* wi = wih_bf + (size_t)l * G_ * H_;
            const __bf16* wh = whh_bf + (size_t)l * G_ * H_;
            float*  hl  = hstate + (size_t)l * nBH;
            __bf16* hlb = h_bf   + (size_t)l * nBH;
            gemm_bf16_kernel<<<gateBlocks, 256, 0, stream>>>(x_bf, wi, nullptr, gi, G_);
            gemm_bf16_kernel<<<gateBlocks, 256, 0, stream>>>(hlb,  wh, nullptr, gh, G_);
            gru_pointwise_kernel<<<bhBlocks, 256, 0, stream>>>(
                gi, gh, b_ih + (size_t)l * G_, b_hh + (size_t)l * G_, hl, hlb, x_bf);
        }
        gemm_bf16_kernel<<<logitsBlocks, 256, 0, stream>>>(
            x_bf, outw_bf, out_b, logits, V_);
        argmax_rows_kernel<<<B_, 256, 0, stream>>>(logits, tok, rowmax);
        logsoftmax_write_kernel<<<B_, 256, 0, stream>>>(logits, rowmax, out, t);
    }

    // h_final appended after log_probs: out[B*T*V ..]
    hipMemcpyAsync(out + (size_t)B_ * T_ * V_, hstate,
                   (size_t)L_ * nBH * sizeof(float),
                   hipMemcpyDeviceToDevice, stream);
}
